// SelfAttention_15126874816540
// MI455X (gfx1250) — compile-verified, matrix operand order checked by reading, not run on hardware
//
#include <hip/hip_runtime.h>

// ---------------------------------------------------------------------------
// MI455X (gfx1250) self-attention-with-patch-embed, bf16x3-split WMMA pipeline
//   s = q @ (kv @ q)   with  kv = Y[b,h,d,p],  q = kv^T
//   (matmul associativity collapses the 1024x1024 attention matrix to a
//    64x64 symmetric Gram matrix: 68.7 GFLOP + 34 GB intermediate -> 4.3 GFLOP)
// Kernel 1: conv patch embed as GEMM  (M=512, N=1024, K=768 per batch)
// Kernel 2: per-(b,h) Gram matrix M64 = Y * Y^T  (64x64, K=1024)
// Kernel 3: s = q * M64               (M=1024, N=64, K=64)
// All matmuls: v_wmma_f32_16x16x32_bf16 with hi/lo bf16 error compensation
// (3 WMMAs per fp32 product => ~2^-16 relative accuracy, near-fp32), since
// the bf16 matrix pipe is ~8x the FLOP/instr of the fp32 WMMA path.
// bf16 fragments are loaded via 16B-run bit-casts (no per-half moves); the
// fp32->hi/lo split is element-wise so clang lowers it to v_fma_mix_f32_bf16.
// ---------------------------------------------------------------------------

typedef __attribute__((ext_vector_type(16))) __bf16        v16bf;
typedef __attribute__((ext_vector_type(8)))  float         v8f;
typedef __attribute__((ext_vector_type(4)))  unsigned int  uv4;
typedef __attribute__((ext_vector_type(4)))  float         fv4;

struct U8 { unsigned int w[8]; };   // 8 dwords == one v16bf fragment
struct P2 { uv4 a, b; };            // two 16B runs == one v16bf fragment

#define WMMA_BF16 __builtin_amdgcn_wmma_f32_16x16x32_bf16

__device__ __forceinline__ unsigned short bf16_trunc_bits(float x) {
  return (unsigned short)(__builtin_bit_cast(unsigned int, x) >> 16);
}
__device__ __forceinline__ float bf16_bits_to_f(unsigned short h) {
  return __builtin_bit_cast(float, ((unsigned int)h) << 16);
}
__device__ __forceinline__ __bf16 bits_to_bf(unsigned short h) {
  return __builtin_bit_cast(__bf16, h);
}
__device__ __forceinline__ v16bf frag_from(const U8& u) {
  return __builtin_bit_cast(v16bf, u);
}
// 16-element bf16 fragment from two contiguous 16-byte runs (pure b128 loads).
__device__ __forceinline__ v16bf load_frag(
    const unsigned short* __restrict__ p0,
    const unsigned short* __restrict__ p1) {
  P2 t;
  t.a = *(const uv4*)p0;
  t.b = *(const uv4*)p1;
  return __builtin_bit_cast(v16bf, t);
}

// Split 8 consecutive fp32 into bf16 hi/lo at fragment elements [BASE..BASE+7].
// Element-wise form: clang lowers the lo path to v_fma_mix_f32_bf16.
template <int BASE>
__device__ __forceinline__ void split8(const float* __restrict__ p,
                                       v16bf& h, v16bf& l) {
  fv4 a = *(const fv4*)p;
  fv4 b = *(const fv4*)(p + 4);
  float v[8] = {a[0], a[1], a[2], a[3], b[0], b[1], b[2], b[3]};
#pragma unroll
  for (int i = 0; i < 8; ++i) {
    const unsigned short hb = bf16_trunc_bits(v[i]);
    h[BASE + i] = bits_to_bf(hb);
    const float lf = v[i] - bf16_bits_to_f(hb);
    l[BASE + i] = bits_to_bf(bf16_trunc_bits(lf));
  }
}

// ---------------------------------------------------------------------------
// Kernel 1: patch-embed GEMM.  Y[b,f,p] = sum_c W[f,c] * X[b,c,p] + bias[f]
// c = ci*256 + ky*16 + kx  (im2col generated on the fly from x).
// Wave tile 32(f) x 64(p); WG = 8 waves => 128x128; grid (4, 8, 32).
// ---------------------------------------------------------------------------
__global__ __launch_bounds__(256) void k_conv_gemm(
    const float* __restrict__ x, const float* __restrict__ w,
    const float* __restrict__ bias, unsigned short* __restrict__ Yhi,
    unsigned short* __restrict__ Ylo) {
  const int lane = threadIdx.x & 31;
  const int wid  = threadIdx.x >> 5;
  const int g    = lane >> 4;   // lane half-group (WMMA layout)
  const int ln   = lane & 15;
  const int b    = blockIdx.z;
  const int f0   = blockIdx.x * 128 + (wid & 3) * 32;
  const int p0   = blockIdx.y * 128 + (wid >> 2) * 64;

  const v8f zero = {0.f, 0.f, 0.f, 0.f, 0.f, 0.f, 0.f, 0.f};
  v8f acc[2][4];
#pragma unroll
  for (int i = 0; i < 2; ++i)
#pragma unroll
    for (int j = 0; j < 4; ++j) acc[i][j] = zero;

  for (int kt = 0; kt < 24; ++kt) {   // K = 768 = 24 * 32
    const int kbase = kt * 32;
    v16bf ahi[2], alo[2];
#pragma unroll
    for (int mi = 0; mi < 2; ++mi) {  // A = conv weights, rows f (contig. K)
      const int f = f0 + mi * 16 + ln;
      const float* wr = w + (size_t)f * 768 + kbase;
      split8<0>(wr + 8 * g, ahi[mi], alo[mi]);
      split8<8>(wr + 16 + 8 * g, ahi[mi], alo[mi]);
    }
#pragma unroll
    for (int ni = 0; ni < 4; ++ni) {  // B = im2col(x), cols p
      const int p  = p0 + ni * 16 + ln;
      const int py = p >> 5, px = p & 31;
      v16bf bhi, blo;
      {
        const int c  = kbase + 8 * g;             // run 0
        const int ci = c >> 8, ky = (c >> 4) & 15, kx = c & 15;
        const float* xp =
            x + (((size_t)(b * 3 + ci) * 512 + py * 16 + ky) * 512 +
                 px * 16 + kx);
        split8<0>(xp, bhi, blo);
      }
      {
        const int c  = kbase + 16 + 8 * g;        // run 1
        const int ci = c >> 8, ky = (c >> 4) & 15, kx = c & 15;
        const float* xp =
            x + (((size_t)(b * 3 + ci) * 512 + py * 16 + ky) * 512 +
                 px * 16 + kx);
        split8<8>(xp, bhi, blo);
      }
#pragma unroll
      for (int mi = 0; mi < 2; ++mi) {  // bf16x3 compensated product
        acc[mi][ni] = WMMA_BF16(false, ahi[mi], false, bhi, (short)0,
                                acc[mi][ni], false, false);
        acc[mi][ni] = WMMA_BF16(false, ahi[mi], false, blo, (short)0,
                                acc[mi][ni], false, false);
        acc[mi][ni] = WMMA_BF16(false, alo[mi], false, bhi, (short)0,
                                acc[mi][ni], false, false);
      }
    }
  }

  // bias + split-store Y (bf16 hi/lo)
#pragma unroll
  for (int mi = 0; mi < 2; ++mi) {
#pragma unroll
    for (int r = 0; r < 8; ++r) {
      const int f = f0 + mi * 16 + r + 8 * g;
      const float bv = bias[f];
#pragma unroll
      for (int ni = 0; ni < 4; ++ni) {
        const int p = p0 + ni * 16 + ln;
        const float v = acc[mi][ni][r] + bv;
        const size_t idx = (((size_t)b * 512 + f) << 10) + p;
        const unsigned short hb = bf16_trunc_bits(v);
        Yhi[idx] = hb;
        Ylo[idx] = bf16_trunc_bits(v - bf16_bits_to_f(hb));
      }
    }
  }
}

// ---------------------------------------------------------------------------
// Kernel 2: per-(b,h) Gram matrix  M64[e,d] = sum_p Y[e,p] * Y[d,p]  (K=1024)
// Note (b*512 + h*64)*1024 == bh*64*1024.  One WG per bh; 8 waves x 2 tiles.
// ---------------------------------------------------------------------------
__global__ __launch_bounds__(256) void k_gram(
    const unsigned short* __restrict__ Yhi,
    const unsigned short* __restrict__ Ylo, unsigned short* __restrict__ Mhi,
    unsigned short* __restrict__ Mlo) {
  const int lane = threadIdx.x & 31;
  const int wid  = threadIdx.x >> 5;
  const int g = lane >> 4, ln = lane & 15;
  const int bh = blockIdx.x;                // 0..255
  const size_t base = (size_t)bh * 64 * 1024;

  const int t0 = wid * 2;                   // two tiles, same row-block
  const int me = (t0 >> 2) * 16;
  const v8f zero = {0.f, 0.f, 0.f, 0.f, 0.f, 0.f, 0.f, 0.f};
  v8f acc[2] = {zero, zero};

  for (int kt = 0; kt < 32; ++kt) {
    const int kb = kt * 32;
    const unsigned short* arh = Yhi + base + (size_t)(me + ln) * 1024 + kb;
    const unsigned short* arl = Ylo + base + (size_t)(me + ln) * 1024 + kb;
    const v16bf ahi = load_frag(arh + 8 * g, arh + 16 + 8 * g);
    const v16bf alo = load_frag(arl + 8 * g, arl + 16 + 8 * g);
#pragma unroll
    for (int j = 0; j < 2; ++j) {
      const int nd = ((t0 + j) & 3) * 16;
      const unsigned short* rh = Yhi + base + (size_t)(nd + ln) * 1024 + kb;
      const unsigned short* rl = Ylo + base + (size_t)(nd + ln) * 1024 + kb;
      const v16bf bhi = load_frag(rh + 8 * g, rh + 16 + 8 * g);
      const v16bf blo = load_frag(rl + 8 * g, rl + 16 + 8 * g);
      acc[j] = WMMA_BF16(false, ahi, false, bhi, (short)0, acc[j], false, false);
      acc[j] = WMMA_BF16(false, ahi, false, blo, (short)0, acc[j], false, false);
      acc[j] = WMMA_BF16(false, alo, false, bhi, (short)0, acc[j], false, false);
    }
  }
#pragma unroll
  for (int j = 0; j < 2; ++j) {
    const int nd = ((t0 + j) & 3) * 16;
#pragma unroll
    for (int r = 0; r < 8; ++r) {
      const int e = me + r + 8 * g;
      const int d = nd + ln;
      const float v = acc[j][r];
      const size_t idx = (size_t)bh * 4096 + e * 64 + d;
      const unsigned short hb = bf16_trunc_bits(v);
      Mhi[idx] = hb;
      Mlo[idx] = bf16_trunc_bits(v - bf16_bits_to_f(hb));
    }
  }
}

// ---------------------------------------------------------------------------
// Kernel 3: s[b,h,p,d] = sum_e q[p,e] * M64[e,d],  q[p,e] = Y[e,p].  K = 64.
// B uses M's symmetry: B[e,d] = M[d][e]  => contiguous row reads.
// Grid (8, 256); wave = 16 p-rows x 64 d cols (4 tiles).
// ---------------------------------------------------------------------------
__global__ __launch_bounds__(256) void k_qm(
    const unsigned short* __restrict__ Yhi,
    const unsigned short* __restrict__ Ylo,
    const unsigned short* __restrict__ Mhi,
    const unsigned short* __restrict__ Mlo, float* __restrict__ out) {
  const int lane = threadIdx.x & 31;
  const int wid  = threadIdx.x >> 5;
  const int g = lane >> 4, ln = lane & 15;
  const int bh   = blockIdx.y;
  const int prow = blockIdx.x * 128 + wid * 16;
  const size_t ybase = (size_t)bh * 64 * 1024;
  const size_t mbase = (size_t)bh * 4096;

  const v8f zero = {0.f, 0.f, 0.f, 0.f, 0.f, 0.f, 0.f, 0.f};
  v8f acc[4] = {zero, zero, zero, zero};

#pragma unroll
  for (int kt = 0; kt < 2; ++kt) {
    const int kb = kt * 32;
    // A[p, e] = Y[e, p]: per-lane strided gathers, packed pairwise into dwords
    U8 ah, al;
#pragma unroll
    for (int j = 0; j < 8; ++j) {
      const int e0 = 2 * j;
      const int K0 = kb + ((e0 >> 3) << 4) + 8 * g + (e0 & 7);
      const size_t i0 = ybase + (size_t)K0 * 1024 + prow + ln;
      const size_t i1 = i0 + 1024;  // element e0+1 -> K0+1 (same run)
      ah.w[j] = (unsigned)Yhi[i0] | ((unsigned)Yhi[i1] << 16);
      al.w[j] = (unsigned)Ylo[i0] | ((unsigned)Ylo[i1] << 16);
    }
    const v16bf ahi = frag_from(ah);
    const v16bf alo = frag_from(al);
#pragma unroll
    for (int ni = 0; ni < 4; ++ni) {
      const unsigned short* rh = Mhi + mbase + (size_t)(ni * 16 + ln) * 64 + kb;
      const unsigned short* rl = Mlo + mbase + (size_t)(ni * 16 + ln) * 64 + kb;
      const v16bf bhi = load_frag(rh + 8 * g, rh + 16 + 8 * g);
      const v16bf blo = load_frag(rl + 8 * g, rl + 16 + 8 * g);
      acc[ni] = WMMA_BF16(false, ahi, false, bhi, (short)0, acc[ni], false, false);
      acc[ni] = WMMA_BF16(false, ahi, false, blo, (short)0, acc[ni], false, false);
      acc[ni] = WMMA_BF16(false, alo, false, bhi, (short)0, acc[ni], false, false);
    }
  }
#pragma unroll
  for (int ni = 0; ni < 4; ++ni) {
#pragma unroll
    for (int r = 0; r < 8; ++r) {
      const int p = prow + r + 8 * g;
      const int d = ni * 16 + ln;
      out[((size_t)bh * 1024 + p) * 64 + d] = acc[ni][r];
    }
  }
}

// ---------------------------------------------------------------------------
extern "C" void kernel_launch(void* const* d_in, const int* in_sizes, int n_in,
                              void* d_out, int out_size, void* d_ws,
                              size_t ws_size, hipStream_t stream) {
  const float* x    = (const float*)d_in[0];  // [32,3,512,512]
  const float* w    = (const float*)d_in[1];  // [512,3,16,16]
  const float* bias = (const float*)d_in[2];  // [512]
  float* out = (float*)d_out;                 // [32,8,1024,64]

  const size_t YN = (size_t)32 * 512 * 1024;  // Y elements
  const size_t MN = (size_t)256 * 64 * 64;    // M elements
  unsigned short* Yhi = (unsigned short*)d_ws;
  unsigned short* Ylo = Yhi + YN;
  unsigned short* Mhi = Ylo + YN;
  unsigned short* Mlo = Mhi + MN;             // total scratch ~68 MB

  k_conv_gemm<<<dim3(4, 8, 32), 256, 0, stream>>>(x, w, bias, Yhi, Ylo);
  k_gram<<<dim3(256), 256, 0, stream>>>(Yhi, Ylo, Mhi, Mlo);
  k_qm<<<dim3(8, 256), 256, 0, stream>>>(Yhi, Ylo, Mhi, Mlo, out);
}
